// ContrastiveModel_55164559949934
// MI455X (gfx1250) — compile-verified, matrix operand order checked by reading, not run on hardware
//
#include <hip/hip_runtime.h>
#include <hip/hip_bf16.h>
#include <math.h>

typedef __attribute__((ext_vector_type(16))) __bf16 v16bf;
typedef __attribute__((ext_vector_type(8)))  float  v8f;
typedef __attribute__((ext_vector_type(4)))  unsigned int u32x4;
typedef __attribute__((ext_vector_type(4)))  int v4i;

#define TAU_F 0.5f
#define LOG2E_F 1.4426950408889634f

#define AS_GLOBAL __attribute__((address_space(1)))
#define AS_LOCAL  __attribute__((address_space(3)))

#if defined(__has_builtin)
#if __has_builtin(__builtin_amdgcn_global_load_async_to_lds_b128)
#define HAVE_ASYNC_COPY 1
#endif
#if __has_builtin(__builtin_amdgcn_s_wait_asynccnt)
#define HAVE_WAIT_ASYNC 1
#endif
#endif

#if defined(HAVE_ASYNC_COPY)
__device__ __forceinline__ void async_cp_b128(const __bf16* g, __bf16* l) {
  const v4i* gv = (const v4i*)g;
  v4i*       lv = (v4i*)l;
  __builtin_amdgcn_global_load_async_to_lds_b128(
      (AS_GLOBAL v4i*)gv, (AS_LOCAL v4i*)lv, 0, 0);
}
__device__ __forceinline__ void wait_async_le4() {
#if defined(HAVE_WAIT_ASYNC)
  __builtin_amdgcn_s_wait_asynccnt(4);
#else
  asm volatile("s_wait_asynccnt 0x4" ::: "memory");
#endif
}
__device__ __forceinline__ void wait_async_0() {
#if defined(HAVE_WAIT_ASYNC)
  __builtin_amdgcn_s_wait_asynccnt(0);
#else
  asm volatile("s_wait_asynccnt 0x0" ::: "memory");
#endif
}
#endif

// ---------------------------------------------------------------------------
// WMMA helpers.
// A-fragment layout (ISA 7.12.2, 16-bit A 16x32): lane = half*16 + (m&15),
//   element j: K = (j/8)*16 + half*8 + (j%8).
// B loaded with the mirrored layout (lane%16 = output column): we feed rows
// of the "B-side" matrix, computing A * B^T — what Z·Zt and X·W^T need.
// ---------------------------------------------------------------------------
__device__ __forceinline__ v16bf frag_bf16(const __bf16* __restrict__ base,
                                           int row0, int ld, int k0, int lane) {
  int r = row0 + (lane & 15);
  int half = (lane >> 4) & 1;
  const __bf16* p = base + (size_t)r * ld + k0 + half * 8;
  union { v16bf v; u32x4 q[2]; } u;
  u.q[0] = *(const u32x4*)(p);        // elements 0..7  : K = half*8 + 0..7
  u.q[1] = *(const u32x4*)(p + 16);   // elements 8..15 : K = 16 + half*8 + 0..7
  return u.v;
}

__device__ __forceinline__ v16bf frag_f32(const float* __restrict__ base,
                                          int row0, int ld, int k0, int lane) {
  int r = row0 + (lane & 15);
  int half = (lane >> 4) & 1;
  const float* p = base + (size_t)r * ld + k0 + half * 8;
  v16bf v;
#pragma unroll
  for (int t = 0; t < 8; ++t) v[t] = (__bf16)p[t];
#pragma unroll
  for (int t = 0; t < 8; ++t) v[8 + t] = (__bf16)p[16 + t];
  return v;
}

__device__ __forceinline__ v8f wmma_bf16(v16bf a, v16bf b, v8f c) {
  return __builtin_amdgcn_wmma_f32_16x16x32_bf16(false, a, false, b,
                                                 (short)0, c, false, false);
}

__device__ __forceinline__ v8f zero8() {
  v8f z;
#pragma unroll
  for (int t = 0; t < 8; ++t) z[t] = 0.0f;
  return z;
}

// ---------------------------------------------------------------------------
// 1) weights f32 -> bf16  (W1, W2 are each 256*256 = 65536 elements)
// ---------------------------------------------------------------------------
__global__ void convert_w_kernel(const float* __restrict__ W1,
                                 const float* __restrict__ W2,
                                 __bf16* __restrict__ W1b,
                                 __bf16* __restrict__ W2b) {
  int i = blockIdx.x * blockDim.x + threadIdx.x;  // 0..131071
  if (i < 65536) W1b[i] = (__bf16)W1[i];
  else           W2b[i - 65536] = (__bf16)W2[i - 65536];
}

__global__ void zero_f32_kernel(float* __restrict__ p, int n) {
  int i = blockIdx.x * blockDim.x + threadIdx.x;
  if (i < n) p[i] = 0.0f;
}

// ---------------------------------------------------------------------------
// 2) Fused projection + row-normalize:
//    Z = normalize( elu(X W1^T + b1) W2^T + b2 )   -> bf16 out
//    Block: 256 thr (8 waves), 64 rows x 256 cols. Wave tile 32x64.
// ---------------------------------------------------------------------------
__global__ void proj_kernel(const float* __restrict__ X1,
                            const float* __restrict__ X2,
                            const __bf16* __restrict__ W1b,
                            const float* __restrict__ b1,
                            const __bf16* __restrict__ W2b,
                            const float* __restrict__ b2,
                            __bf16* __restrict__ Z1b,
                            __bf16* __restrict__ Z2b) {
  __shared__ __bf16 Hs[64 * 256];   // elu output, bf16 (32 KB)
  __shared__ float  normsq[64];

  const float* X  = blockIdx.y ? X2  : X1;
  __bf16*      Zb = blockIdx.y ? Z2b : Z1b;
  const int rowBase = blockIdx.x * 64;

  const int lane = threadIdx.x & 31;
  const int wave = threadIdx.x >> 5;
  const int wm = wave & 1;   // M: 32*wm
  const int wn = wave >> 1;  // N: 64*wn
  const int half = lane >> 4, l15 = lane & 15;

  v8f acc[2][4];
#pragma unroll
  for (int i = 0; i < 2; ++i)
#pragma unroll
    for (int j = 0; j < 4; ++j) acc[i][j] = zero8();

  // ---- stage 1: H = elu(X W1^T + b1) ----
  for (int ks = 0; ks < 256; ks += 32) {
    v16bf a[2], bfr[4];
#pragma unroll
    for (int i = 0; i < 2; ++i)
      a[i] = frag_f32(X, rowBase + wm * 32 + 16 * i, 256, ks, lane);
#pragma unroll
    for (int j = 0; j < 4; ++j)
      bfr[j] = frag_bf16(W1b, wn * 64 + 16 * j, 256, ks, lane);
#pragma unroll
    for (int i = 0; i < 2; ++i)
#pragma unroll
      for (int j = 0; j < 4; ++j)
        acc[i][j] = wmma_bf16(a[i], bfr[j], acc[i][j]);
  }

#pragma unroll
  for (int j = 0; j < 4; ++j) {
    int p = wn * 64 + 16 * j + l15;
    float bj = b1[p];
#pragma unroll
    for (int i = 0; i < 2; ++i)
#pragma unroll
      for (int k = 0; k < 8; ++k) {
        float h = acc[i][j][k] + bj;
        h = h > 0.0f ? h : (expf(h) - 1.0f);   // elu
        int r = wm * 32 + 16 * i + k + 8 * half;
        Hs[r * 256 + p] = (__bf16)h;
      }
  }
  __syncthreads();

  // ---- stage 2: Z = H W2^T + b2 ----
#pragma unroll
  for (int i = 0; i < 2; ++i)
#pragma unroll
    for (int j = 0; j < 4; ++j) acc[i][j] = zero8();

  for (int ks = 0; ks < 256; ks += 32) {
    v16bf a[2], bfr[4];
#pragma unroll
    for (int i = 0; i < 2; ++i)
      a[i] = frag_bf16(Hs, wm * 32 + 16 * i, 256, ks, lane);
#pragma unroll
    for (int j = 0; j < 4; ++j)
      bfr[j] = frag_bf16(W2b, wn * 64 + 16 * j, 256, ks, lane);
#pragma unroll
    for (int i = 0; i < 2; ++i)
#pragma unroll
      for (int j = 0; j < 4; ++j)
        acc[i][j] = wmma_bf16(a[i], bfr[j], acc[i][j]);
  }

  if (threadIdx.x < 64) normsq[threadIdx.x] = 0.0f;
  __syncthreads();

  // bias + accumulate row sum-of-squares (LDS atomics)
#pragma unroll
  for (int j = 0; j < 4; ++j) {
    float bj = b2[wn * 64 + 16 * j + l15];
#pragma unroll
    for (int i = 0; i < 2; ++i)
#pragma unroll
      for (int k = 0; k < 8; ++k)
        acc[i][j][k] += bj;
  }
#pragma unroll
  for (int i = 0; i < 2; ++i)
#pragma unroll
    for (int k = 0; k < 8; ++k) {
      float s = 0.0f;
#pragma unroll
      for (int j = 0; j < 4; ++j) { float z = acc[i][j][k]; s += z * z; }
      atomicAdd(&normsq[wm * 32 + 16 * i + k + 8 * half], s);
    }
  __syncthreads();

  // normalize + store bf16
#pragma unroll
  for (int i = 0; i < 2; ++i)
#pragma unroll
    for (int k = 0; k < 8; ++k) {
      int r = wm * 32 + 16 * i + k + 8 * half;
      float inv = 1.0f / fmaxf(sqrtf(normsq[r]), 1e-12f);
      size_t grow = (size_t)(rowBase + r) * 256;
#pragma unroll
      for (int j = 0; j < 4; ++j)
        Zb[grow + wn * 64 + 16 * j + l15] = (__bf16)(acc[i][j][k] * inv);
    }
}

// ---------------------------------------------------------------------------
// 3) diagonal dots per row: d12 = z1.z2, d11 = z1.z1, d22 = z2.z2 (f32 accum)
// ---------------------------------------------------------------------------
__global__ void diag_kernel(const __bf16* __restrict__ z1,
                            const __bf16* __restrict__ z2,
                            float* __restrict__ d12,
                            float* __restrict__ d11,
                            float* __restrict__ d22) {
  int wave = threadIdx.x >> 5, lane = threadIdx.x & 31;
  int row = blockIdx.x * 8 + wave;
  const __bf16* p1 = z1 + (size_t)row * 256;
  const __bf16* p2 = z2 + (size_t)row * 256;
  float s12 = 0.0f, s11 = 0.0f, s22 = 0.0f;
#pragma unroll
  for (int t = 0; t < 8; ++t) {
    int c = lane + 32 * t;
    float a = (float)p1[c], b = (float)p2[c];
    s12 += a * b; s11 += a * a; s22 += b * b;
  }
#pragma unroll
  for (int m = 1; m < 32; m <<= 1) {
    s12 += __shfl_xor(s12, m, 32);
    s11 += __shfl_xor(s11, m, 32);
    s22 += __shfl_xor(s22, m, 32);
  }
  if (lane == 0) { d12[row] = s12; d11[row] = s11; d22[row] = s22; }
}

// ---------------------------------------------------------------------------
// 4) Similarity tile kernel: S = ZA ZB^T (per batch), fused exp(./tau) and
//    row/col-sum reduction. Block = 128x128 tile, wave = 64x32 (4x2 WMMA).
//    K-slices of A and B are double-buffered in LDS via async global->LDS
//    copies (ASYNCcnt) so slice s+1 streams in while slice s is in the WMMAs.
//    LDS rows padded to 40 bf16 (80 B) for bank spread; 16B-aligned b128 ops.
// ---------------------------------------------------------------------------
#define SIM_LDSTRIDE 40   // elements per 32-col row (32 data + 8 pad)

__global__ void sim_kernel(const __bf16* __restrict__ ZA,
                           const __bf16* __restrict__ ZB,
                           float* __restrict__ rowsum,
                           float* __restrict__ colsum,   // may be null
                           int L) {
  __shared__ __bf16 As[2][128 * SIM_LDSTRIDE];
  __shared__ __bf16 Bs[2][128 * SIM_LDSTRIDE];

  const int b = blockIdx.z;
  const __bf16* A  = ZA + (size_t)b * L * 256;
  const __bf16* Bm = ZB + (size_t)b * L * 256;
  float* rs = rowsum + (size_t)b * L;
  float* cs = colsum ? colsum + (size_t)b * L : nullptr;

  const int lane = threadIdx.x & 31;
  const int wave = threadIdx.x >> 5;
  const int wm = wave & 1, wn = wave >> 1;
  const int RMb = blockIdx.y * 128;
  const int CNb = blockIdx.x * 128;
  const int RM = RMb + wm * 64;   // wave row base (global)
  const int CN = CNb + wn * 32;   // wave col base (global)
  const int half = lane >> 4, l15 = lane & 15;

  // staging: 128 rows x 32 cols bf16 = 512 x 16B chunks per matrix;
  // thread owns chunks tid and tid+256. chunk c -> row c>>2, 16B-seg c&3.
  const int cA0 = threadIdx.x, cA1 = threadIdx.x + 256;
  const int rA0 = cA0 >> 2, sA0 = cA0 & 3;
  const int rA1 = cA1 >> 2, sA1 = cA1 & 3;

  auto stage = [&](int buf, int ks) {
    const __bf16* gA0 = A  + (size_t)(RMb + rA0) * 256 + ks + sA0 * 8;
    const __bf16* gA1 = A  + (size_t)(RMb + rA1) * 256 + ks + sA1 * 8;
    const __bf16* gB0 = Bm + (size_t)(CNb + rA0) * 256 + ks + sA0 * 8;
    const __bf16* gB1 = Bm + (size_t)(CNb + rA1) * 256 + ks + sA1 * 8;
    __bf16* lA0 = &As[buf][rA0 * SIM_LDSTRIDE + sA0 * 8];
    __bf16* lA1 = &As[buf][rA1 * SIM_LDSTRIDE + sA1 * 8];
    __bf16* lB0 = &Bs[buf][rA0 * SIM_LDSTRIDE + sA0 * 8];
    __bf16* lB1 = &Bs[buf][rA1 * SIM_LDSTRIDE + sA1 * 8];
#if defined(HAVE_ASYNC_COPY)
    async_cp_b128(gA0, lA0);
    async_cp_b128(gA1, lA1);
    async_cp_b128(gB0, lB0);
    async_cp_b128(gB1, lB1);
#else
    *(u32x4*)lA0 = *(const u32x4*)gA0;
    *(u32x4*)lA1 = *(const u32x4*)gA1;
    *(u32x4*)lB0 = *(const u32x4*)gB0;
    *(u32x4*)lB1 = *(const u32x4*)gB1;
#endif
  };

  v8f acc[4][2];
#pragma unroll
  for (int i = 0; i < 4; ++i)
#pragma unroll
    for (int j = 0; j < 2; ++j) acc[i][j] = zero8();

  stage(0, 0);
  for (int s = 0; s < 8; ++s) {
    const int buf = s & 1;
    if (s + 1 < 8) stage(buf ^ 1, (s + 1) * 32);   // overlap copy with math
#if defined(HAVE_ASYNC_COPY)
    if (s + 1 < 8) wait_async_le4();               // retire stage s only
    else           wait_async_0();
#endif
    __syncthreads();                               // slice s visible to all

    v16bf a[4], bq[2];
#pragma unroll
    for (int i = 0; i < 4; ++i)
      a[i] = frag_bf16(&As[buf][0], wm * 64 + 16 * i, SIM_LDSTRIDE, 0, lane);
#pragma unroll
    for (int j = 0; j < 2; ++j)
      bq[j] = frag_bf16(&Bs[buf][0], wn * 32 + 16 * j, SIM_LDSTRIDE, 0, lane);
#pragma unroll
    for (int i = 0; i < 4; ++i)
#pragma unroll
      for (int j = 0; j < 2; ++j)
        acc[i][j] = wmma_bf16(a[i], bq[j], acc[i][j]);

    __syncthreads();                               // done reading buf
  }

  const float S = (1.0f / TAU_F) * LOG2E_F;   // exp(x/tau) = exp2(x*S)
  float colacc[2] = {0.0f, 0.0f};

#pragma unroll
  for (int i = 0; i < 4; ++i) {
    float rowacc[8];
#pragma unroll
    for (int k = 0; k < 8; ++k) rowacc[k] = 0.0f;
#pragma unroll
    for (int j = 0; j < 2; ++j)
#pragma unroll
      for (int k = 0; k < 8; ++k) {
        float e = exp2f(acc[i][j][k] * S);
        rowacc[k] += e;
        colacc[j] += e;
      }
    // reduce across the 16 lanes holding the same output row
#pragma unroll
    for (int k = 0; k < 8; ++k) {
      float t = rowacc[k];
      t += __shfl_xor(t, 1, 32);
      t += __shfl_xor(t, 2, 32);
      t += __shfl_xor(t, 4, 32);
      t += __shfl_xor(t, 8, 32);
      if (l15 == 0)
        atomicAdd(&rs[RM + 16 * i + k + 8 * half], t);
    }
  }

  if (cs) {
#pragma unroll
    for (int j = 0; j < 2; ++j) {
      float u = colacc[j] + __shfl_xor(colacc[j], 16, 32);
      if (half == 0)
        atomicAdd(&cs[CN + 16 * j + l15], u);
    }
  }
}

// ---------------------------------------------------------------------------
// 5) final scalar:
//    out = (1/B) * sum_{b,l} [ -d12/tau + 0.5*(log den1 + log den2) ]
//    den1 = rs11 + rs12 - exp(d11/tau),  den2 = rs22 + cs12 - exp(d22/tau)
// ---------------------------------------------------------------------------
__global__ void loss_kernel(const float* __restrict__ d12,
                            const float* __restrict__ d11,
                            const float* __restrict__ d22,
                            const float* __restrict__ rs1,
                            const float* __restrict__ rs2,
                            const float* __restrict__ rs12,
                            const float* __restrict__ cs12,
                            float* __restrict__ out) {
  __shared__ float red[256];
  const float invTau = 1.0f / TAU_F;
  float acc = 0.0f;
  for (int i = threadIdx.x; i < 16384; i += 256) {
    float den1 = rs1[i] + rs12[i] - expf(invTau * d11[i]);
    float den2 = rs2[i] + cs12[i] - expf(invTau * d22[i]);
    acc += -invTau * d12[i] + 0.5f * (logf(den1) + logf(den2));
  }
  red[threadIdx.x] = acc;
  __syncthreads();
  for (int s = 128; s > 0; s >>= 1) {
    if (threadIdx.x < s) red[threadIdx.x] += red[threadIdx.x + s];
    __syncthreads();
  }
  if (threadIdx.x == 0) out[0] = red[0] * 0.25f;   // mean over B=4
}

// ---------------------------------------------------------------------------
extern "C" void kernel_launch(void* const* d_in, const int* in_sizes, int n_in,
                              void* d_out, int out_size, void* d_ws, size_t ws_size,
                              hipStream_t stream) {
  const float* X1 = (const float*)d_in[0];
  const float* X2 = (const float*)d_in[1];
  const float* W1 = (const float*)d_in[2];
  const float* b1 = (const float*)d_in[3];
  const float* W2 = (const float*)d_in[4];
  const float* b2 = (const float*)d_in[5];
  float* out = (float*)d_out;

  // workspace layout (~17 MB total)
  char* ws = (char*)d_ws;
  __bf16* z1b = (__bf16*)ws;                                   // 8 MB
  __bf16* z2b = z1b + (size_t)16384 * 256;                     // 8 MB
  __bf16* W1b = (__bf16*)(ws + (size_t)16 * 1024 * 1024);      // 128 KB
  __bf16* W2b = W1b + 65536;                                   // 128 KB
  float* fb   = (float*)(ws + (size_t)16 * 1024 * 1024 + 262144);
  float* d12  = fb;
  float* d11  = fb + 16384;
  float* d22  = fb + 2 * 16384;
  float* rs1  = fb + 3 * 16384;   // rs1, rs2, rs12, cs12 contiguous (zeroed)
  float* rs2  = fb + 4 * 16384;
  float* rs12 = fb + 5 * 16384;
  float* cs12 = fb + 6 * 16384;

  convert_w_kernel<<<512, 256, 0, stream>>>(W1, W2, W1b, W2b);
  zero_f32_kernel<<<256, 256, 0, stream>>>(rs1, 4 * 16384);
  proj_kernel<<<dim3(256, 2), 256, 0, stream>>>(X1, X2, W1b, b1, W2b, b2, z1b, z2b);
  diag_kernel<<<2048, 256, 0, stream>>>(z1b, z2b, d12, d11, d22);
  sim_kernel<<<dim3(32, 32, 4), 256, 0, stream>>>(z1b, z1b, rs1,  nullptr, 4096);
  sim_kernel<<<dim3(32, 32, 4), 256, 0, stream>>>(z2b, z2b, rs2,  nullptr, 4096);
  sim_kernel<<<dim3(32, 32, 4), 256, 0, stream>>>(z1b, z2b, rs12, cs12,    4096);
  loss_kernel<<<1, 256, 0, stream>>>(d12, d11, d22, rs1, rs2, rs12, cs12, out);
}
